// PEAJKBaseChannel_73358041416345
// MI455X (gfx1250) — compile-verified, matrix operand order checked by reading, not run on hardware
//
#include <hip/hip_runtime.h>
#include <hip/hip_bf16.h>

typedef __attribute__((ext_vector_type(16))) __bf16 v16bf;
typedef __attribute__((ext_vector_type(8)))  __bf16 v8bf;
typedef __attribute__((ext_vector_type(8)))  float  v8f;

#define DFEAT 128

// ---------------------------------------------------------------- fill
__global__ void fill_kernel(float* p, long long n, float v) {
    long long i = (long long)blockIdx.x * blockDim.x + threadIdx.x;
    if (i < n) p[i] = v;
}

// ---------------------------------------------------------------- fp32 -> bf16 convert (for x)
__global__ void cvt_bf16_kernel(const float* __restrict__ in, __bf16* __restrict__ out,
                                long long n) {
    long long i = (long long)blockIdx.x * blockDim.x + threadIdx.x;
    if (i < n) out[i] = (__bf16)in[i];
}

// ---------------------------------------------------------------- degree count (self loop handled by init=1)
__global__ void deg_count_kernel(const int* dst, float* deg, int E) {
    int e = blockIdx.x * blockDim.x + threadIdx.x;
    if (e < E) atomicAdd(&deg[dst[e]], 1.0f);
}

__global__ void rsqrt_kernel(float* d, int n) {
    int i = blockIdx.x * blockDim.x + threadIdx.x;
    if (i < n) d[i] = rsqrtf(d[i]);   // deg >= 1 always (self loops)
}

// ---------------------------------------------------------------- pack weights (K x cols fp32, row-major, ldb==cols)
// into bf16 WMMA lane order: Bp[((ct*KS + ks)*32 + lane)*16 + j] so each lane's
// whole 16-element B fragment is one contiguous 32-byte chunk.
__global__ void pack_b_kernel(const float* __restrict__ B, __bf16* __restrict__ Bp,
                              int K, int cols) {
    const int idx = blockIdx.x * blockDim.x + threadIdx.x;
    if (idx >= cols * K) return;
    const int j    = idx & 15;
    const int l    = (idx >> 4) & 31;
    const int rest = idx >> 9;           // ct*KS + ks
    const int KS   = K >> 5;
    const int ct   = rest / KS;
    const int ks   = rest - ct * KS;
    const int half = l >> 4;
    const int col  = ct * 16 + (l & 15);
    const int v    = j >> 1;
    // ISA 16-bit fragment k-map: v<4 -> 2v, v>=4 -> 16+2(v-4); +8 for lane half 1
    const int kb = (v < 4 ? 2 * v : 16 + 2 * (v - 4)) + 8 * half + (j & 1) + ks * 32;
    Bp[idx] = (__bf16)B[(size_t)kb * cols + col];
}

// ---------------------------------------------------------------- WMMA GEMM: C[nrows x cols] = A[nrows x K](bf16) * Bpack (+bias)
// One wave -> one 16x16 tile. blockDim.x = 32*(cols/16). gridDim.x = nrows/16 (nrows % 16 == 0).
// The 16 x K bf16 A tile is staged once per block into LDS via async global->LDS copies
// (ASYNCcnt), then all waves read their fragments with ds_load_b128.
template <int K, bool HAS_BIAS>
__global__ void wmma_gemm_kernel(const __bf16* __restrict__ A, int lda,
                                 const __bf16* __restrict__ Bp,
                                 const float* __restrict__ bias,
                                 float* __restrict__ C, int ldc) {
    extern __shared__ __bf16 tile[];          // [16][K] bf16, only LDS object -> offset 0
    const int rowbase = blockIdx.x * 16;
    const int wv      = threadIdx.x >> 5;     // col tile
    const int lane    = threadIdx.x & 31;
    const int half    = lane >> 4;
    const int mr      = lane & 15;
    const int col     = wv * 16 + mr;
    constexpr int KS  = K / 32;
    constexpr int CPR = K / 8;                // 16B chunks per row

    // ---- async stage A tile: 16 rows x K bf16 = 2K 16-byte chunks
    for (int c = threadIdx.x; c < 2 * K; c += blockDim.x) {
        const int r     = c / CPR;
        const int inrow = c - r * CPR;
        const __bf16* gp = A + (size_t)(rowbase + r) * lda + inrow * 8;
        const unsigned ldsaddr = (unsigned)(uintptr_t)(tile + c * 8);
        asm volatile("global_load_async_to_lds_b128 %0, %1, off"
                     :: "v"(ldsaddr), "v"(gp) : "memory");
    }
    asm volatile("s_wait_asynccnt 0x0" ::: "memory");
    __syncthreads();

    v8f c = {};
#pragma unroll
    for (int ks = 0; ks < KS; ++ks) {
        const int kk8 = ks * 32 + 8 * half;
        // A fragment: two ds_load_b128 from the staged tile
        const v8bf a0 = *(const v8bf*)(tile + mr * K + kk8);
        const v8bf a1 = *(const v8bf*)(tile + mr * K + kk8 + 16);
        const v16bf a = __builtin_shufflevector(a0, a1, 0, 1, 2, 3, 4, 5, 6, 7,
                                                8, 9, 10, 11, 12, 13, 14, 15);
        // B fragment: one contiguous 32B chunk per lane (L1-resident packed weights)
        const v16bf b = *(const v16bf*)(Bp + (((size_t)wv * KS + ks) * 32 + lane) * 16);

        c = __builtin_amdgcn_wmma_f32_16x16x32_bf16(
                /*neg_a=*/false, a, /*neg_b=*/false, b,
                /*c_mod=*/(short)0, c, /*reuse_a=*/false, /*reuse_b=*/false);
    }

    float bv = 0.0f;
    if (HAS_BIAS) bv = bias[col];
#pragma unroll
    for (int r = 0; r < 8; ++r) {
        // C/D layout: M = r + 8*half, N = lane&15
        C[(size_t)(rowbase + r + 8 * half) * ldc + col] = c[r] + bv;
    }
}

// ---------------------------------------------------------------- edge scatter: one wave per (edge or self loop)
// lane L handles features [4L, 4L+4); 4 fp32 global atomics into agg[dst].
__global__ void scatter_kernel(const float* __restrict__ t,
                               const int* __restrict__ src,
                               const int* __restrict__ dst,
                               const float* __restrict__ dis,
                               float* __restrict__ agg,
                               int E, int N) {
    const int wavesPerBlk = blockDim.x >> 5;
    const long long w = (long long)blockIdx.x * wavesPerBlk + (threadIdx.x >> 5);
    const long long total = (long long)E + N;
    if (w >= total) return;
    const int lane = threadIdx.x & 31;

    int s, d;
    if (w < E) { s = src[w]; d = dst[w]; }
    else       { s = d = (int)(w - E); }             // self loop
    const float nw = dis[s] * dis[d];

    const float4 v = *(const float4*)(t + (size_t)s * DFEAT + lane * 4);
    float* out = agg + (size_t)d * DFEAT + lane * 4;
    atomicAdd(out + 0, v.x * nw);
    atomicAdd(out + 1, v.y * nw);
    atomicAdd(out + 2, v.z * nw);
    atomicAdd(out + 3, v.w * nw);
}

// ---------------------------------------------------------------- bias + relu -> bf16 layer slice of N x 384 concat buffer
__global__ void bias_relu_kernel(const float* __restrict__ agg,
                                 const float* __restrict__ bias,
                                 __bf16* __restrict__ hdst,  // hcat_bf + layer*128, row stride ldh
                                 int N, int ldh) {
    long long i = (long long)blockIdx.x * blockDim.x + threadIdx.x;
    if (i >= (long long)N * DFEAT) return;
    const int n = (int)(i >> 7);
    const int f = (int)(i & (DFEAT - 1));
    const float v = agg[i] + bias[f];
    hdst[(size_t)n * ldh + f] = (__bf16)(v > 0.0f ? v : 0.0f);
}

// ---------------------------------------------------------------- launcher
extern "C" void kernel_launch(void* const* d_in, const int* in_sizes, int n_in,
                              void* d_out, int out_size, void* d_ws, size_t ws_size,
                              hipStream_t stream) {
    const float* x     = (const float*)d_in[0];
    const int*   edge  = (const int*)d_in[1];
    const float* Ws    = (const float*)d_in[2];
    const float* bs    = (const float*)d_in[3];
    const float* lin_w = (const float*)d_in[4];
    const float* lin_b = (const float*)d_in[5];

    const int H  = DFEAT;                  // 128
    const int N  = in_sizes[0] / H;        // 100000 (multiple of 16)
    const int E  = in_sizes[1] / 2;        // 1600000
    const int S  = in_sizes[2] / (H * H);  // 3
    const int R  = in_sizes[5];            // 64
    const int HC = S * H;                  // 384

    const int* src = edge;
    const int* dst = edge + E;

    // workspace layout (floats), 256B-aligned slices
    auto align64 = [](size_t v) { return (v + 63) & ~(size_t)63; };
    float*  wsf    = (float*)d_ws;
    size_t  off    = 0;
    float*  dis    = wsf + off; off += align64((size_t)N);
    float*  t      = wsf + off; off += align64((size_t)N * H);
    float*  agg    = wsf + off; off += align64((size_t)N * H);
    __bf16* hcatbf = (__bf16*)(wsf + off); off += align64((size_t)N * HC / 2);
    __bf16* xbf    = (__bf16*)(wsf + off); off += align64((size_t)N * H / 2);
    __bf16* bpck   = (__bf16*)(wsf + off); off += align64((size_t)(HC * R + 1) / 2);
    (void)ws_size; (void)n_in; (void)out_size;

    const int TPB = 256;
    const long long nh = (long long)N * H;
    const int nhBlocks = (int)((nh + TPB - 1) / TPB);

    // 0) x -> bf16 once (all GEMM A-operands are bf16 in memory)
    cvt_bf16_kernel<<<nhBlocks, TPB, 0, stream>>>(x, xbf, nh);

    // 1) symmetric degree norm: deg = 1 (self loop) + in-degree; dis = rsqrt(deg)
    fill_kernel<<<(N + TPB - 1) / TPB, TPB, 0, stream>>>(dis, N, 1.0f);
    deg_count_kernel<<<(E + TPB - 1) / TPB, TPB, 0, stream>>>(dst, dis, E);
    rsqrt_kernel<<<(N + TPB - 1) / TPB, TPB, 0, stream>>>(dis, N);

    // 2) GCN layers
    const long long totMsg = (long long)E + N;
    const int scatterBlocks = (int)((totMsg + 7) / 8);           // 8 waves / block
    const int rowTiles = N / 16;

    for (int i = 0; i < S; ++i) {
        const __bf16* A   = (i == 0) ? xbf : (hcatbf + (size_t)(i - 1) * H);
        const int     lda = (i == 0) ? H : HC;

        // pack Ws[i] (128x128) into WMMA lane order, then t = h @ Ws[i]
        pack_b_kernel<<<(H * H + TPB - 1) / TPB, TPB, 0, stream>>>(
            Ws + (size_t)i * H * H, bpck, H, H);
        wmma_gemm_kernel<128, false>
            <<<rowTiles, 32 * (H / 16), 16 * 128 * sizeof(__bf16), stream>>>(
                A, lda, bpck, nullptr, t, H);

        fill_kernel<<<nhBlocks, TPB, 0, stream>>>(agg, nh, 0.0f);

        scatter_kernel<<<scatterBlocks, TPB, 0, stream>>>(t, src, dst, dis, agg, E, N);

        bias_relu_kernel<<<nhBlocks, TPB, 0, stream>>>(
            agg, bs + (size_t)i * H, hcatbf + (size_t)i * H, N, HC);
    }

    // 3) JumpingKnowledge cat + final linear: out = hcat @ lin_w + lin_b (K=384, 64 cols -> 4 waves)
    pack_b_kernel<<<(HC * R + TPB - 1) / TPB, TPB, 0, stream>>>(lin_w, bpck, HC, R);
    wmma_gemm_kernel<384, true>
        <<<rowTiles, 32 * (R / 16), 16 * 384 * sizeof(__bf16), stream>>>(
            hcatbf, HC, bpck, lin_b, (float*)d_out, R);
}
